// CustomRNNCell_71236327571901
// MI455X (gfx1250) — matrix operand order repaired and checked by reading, not compiled, not run on hardware
//
#include <hip/hip_runtime.h>
#include <cstdint>

// ---------------- problem dims ----------------
#define BB 64
#define TT 512
#define II 1024
#define HH 1024

#define KT   32   // K-step per WMMA pass
#define ROWP 40   // padded LDS row pitch (bf16 elems): 80B -> 16B aligned, bank-spread

typedef __bf16 bf16_t;
typedef bf16_t v16bf __attribute__((ext_vector_type(16)));
typedef bf16_t v8bf  __attribute__((ext_vector_type(8)));
typedef float  v8f   __attribute__((ext_vector_type(8)));
typedef unsigned int u32x4 __attribute__((ext_vector_type(4)));
typedef int          i32x8 __attribute__((ext_vector_type(8)));
typedef int          i32x4 __attribute__((ext_vector_type(4)));

#define SHUF16(a, b) __builtin_shufflevector((a), (b), 0,1,2,3,4,5,6,7,8,9,10,11,12,13,14,15)

__device__ __forceinline__ uint16_t f32_to_bf16_rne(float f) {
  uint32_t u = __float_as_uint(f);
  uint32_t r = (u >> 16) & 1u;
  return (uint16_t)((u + 0x7FFFu + r) >> 16);
}
__device__ __forceinline__ float bf16_to_f32(uint16_t u) {
  return __uint_as_float(((uint32_t)u) << 16);
}

// ---------------- elementwise helpers ----------------
__global__ __launch_bounds__(256) void cvt_f32_bf16_kernel(const float* __restrict__ src,
                                                           uint16_t* __restrict__ dst, int n) {
  int i = blockIdx.x * 256 + threadIdx.x;
  if (i < n) dst[i] = f32_to_bf16_rne(src[i]);
}
__global__ __launch_bounds__(256) void zero_f32_kernel(float* __restrict__ p, int n) {
  int i = blockIdx.x * 256 + threadIdx.x;
  if (i < n) p[i] = 0.0f;
}

// ---------------- TDM: load a rows x 32 bf16 tile (row stride 1024 elems) into padded LDS ----
// D# per CDNA5 ISA ch.8: group0 = {flags, lds_addr, global_addr lo, global_addr hi | type=2}
// group1 packs data_size=2B, pad_enable (interval 16 DW, amount 4 DW -> ROWP pitch),
// tensor_dim0/1 = 1024, tile_dim0=32, tile_dim1=rows, tensor_dim0_stride = 1024.
__device__ __forceinline__ void tdm_load_w_tile(const uint16_t* gsrc, uint32_t lds_off, int rows) {
  uint64_t ga = (uint64_t)(uintptr_t)gsrc;
  u32x4 g0;
  g0[0] = 0x1u;                                               // count=1, user descriptor
  g0[1] = lds_off;                                            // lds_addr (bytes)
  g0[2] = (uint32_t)ga;                                       // global_addr[31:0]
  g0[3] = (uint32_t)((ga >> 32) & 0x01FFFFFFu) | (2u << 30);  // addr[56:32] | type=2
  i32x8 g1;
  g1[0] = (int)((1u << 16) |                 // data_size = 1 -> 2 bytes
                (1u << 20) |                 // pad_enable
                (3u << 22) |                 // pad_interval: 16 DWORDs (64B row)
                (3u << 25));                 // pad_amount: 4 DWORDs (16B pad)
  g1[1] = (int)(((uint32_t)(HH & 0xFFFF)) << 16);            // tensor_dim0 lo16 (bits 63:48)
  g1[2] = (int)(((uint32_t)(HH & 0xFFFF)) << 16);            // dim0 hi16=0 | tensor_dim1 lo16
  g1[3] = (int)(((uint32_t)KT) << 16);                       // dim1 hi16=0 | tile_dim0=32
  g1[4] = (int)((uint32_t)rows & 0xFFFFu);                   // tile_dim1=rows, tile_dim2=0
  g1[5] = (int)HH;                                           // tensor_dim0_stride lo32
  g1[6] = 0;
  g1[7] = 0;
  i32x4 z4 = {0, 0, 0, 0};
#if __has_include(<hip/amd_detail/amd_gfx1250_TDM.h>)
  i32x8 z8 = {0, 0, 0, 0, 0, 0, 0, 0};
  __builtin_amdgcn_tensor_load_to_lds(g0, g1, z4, z4, z8, 0);
#else
  __builtin_amdgcn_tensor_load_to_lds(g0, g1, z4, z4, 0);
#endif
}

// =====================================================================================
// Shared tile geometry: 64(M) x 128(N) macro-tile, 8 waves in a 2x4 grid,
// each wave owns a 2x2 grid of 16x16 f32 accumulators (4 v_wmma per K-step).
// A (activations, f32 in global) is staged+converted by all threads.
// B (weights, bf16 K-major) is staged by the Tensor Data Mover.
// =====================================================================================

// =====================================================================================
// Phase 1: xp[m][n] = bf16( x[m][:] . W_x[n][:] + b_x[n] + b_h[n] ),  m = b*T + t
// M = B*T = 32768, N = H = 1024, K = I = 1024.
// =====================================================================================
__global__ __launch_bounds__(256)
void xproj_kernel(const float* __restrict__ x,      // [M,K] f32
                  const uint16_t* __restrict__ wxb, // [N,K] bf16
                  const float* __restrict__ bx,
                  const float* __restrict__ bh,
                  uint16_t* __restrict__ xp)        // [M,N] bf16
{
  __shared__ uint16_t As[64 * ROWP];
  __shared__ uint16_t Bs[128 * ROWP];

  const int m0   = blockIdx.y * 64;
  const int n0   = blockIdx.x * 128;
  const int tid  = threadIdx.x;
  const int lane = tid & 31;
  const int wv   = tid >> 5;
  const int wm   = (wv & 1) * 2;   // m fragment base (0 or 2)
  const int wn   = (wv >> 1) * 2;  // n fragment base (0,2,4,6)

  const int srow = tid >> 2;       // A staging: 64 rows x 8 elems
  const int scol = (tid & 3) * 8;

  const int half = lane >> 4;
  const int r16  = lane & 15;

  const uint32_t bs_lds_off = (uint32_t)(uintptr_t)(void*)Bs;

  v8f acc[2][2] = {};

  for (int k0 = 0; k0 < II; k0 += KT) {
    __syncthreads();
    if (wv == 0) {  // TDM pulls the 128x32 W_x tile into padded LDS (EXEC ignored by TDM)
      tdm_load_w_tile(wxb + (size_t)n0 * II + k0, bs_lds_off, 128);
    }
    { // stage A: x (f32 -> bf16), 64 x 32 tile
      const float* gp = x + (size_t)(m0 + srow) * II + (k0 + scol);
      uint16_t*    lp = As + srow * ROWP + scol;
      __builtin_prefetch((const void*)(gp + KT), 0, 3);
      #pragma unroll
      for (int j = 0; j < 8; ++j) lp[j] = f32_to_bf16_rne(gp[j]);
    }
    if (wv == 0) __builtin_amdgcn_s_wait_tensorcnt(0);
    __syncthreads();

    // A fragments: lane -> row, K = kbase..kbase+7 then kbase+16..+23 (kbase = half*8)
    v16bf afrag[2];
    #pragma unroll
    for (int jm = 0; jm < 2; ++jm) {
      const uint16_t* ap = As + ((wm + jm) * 16 + r16) * ROWP + half * 8;
      v8bf a_lo = *(const v8bf*)(const void*)(ap);
      v8bf a_hi = *(const v8bf*)(const void*)(ap + 16);
      afrag[jm] = SHUF16(a_lo, a_hi);
    }
    #pragma unroll
    for (int jn = 0; jn < 2; ++jn) {
      // B fragment: lane -> column n, K = half*8..half*8+7 then half*8+16..+23
      const uint16_t* bp = Bs + ((wn + jn) * 16 + r16) * ROWP + half * 8;
      v8bf b_lo = *(const v8bf*)(const void*)(bp);
      v8bf b_hi = *(const v8bf*)(const void*)(bp + 16);
      v16bf bfrag = SHUF16(b_lo, b_hi);
      #pragma unroll
      for (int jm = 0; jm < 2; ++jm)
        acc[jm][jn] = __builtin_amdgcn_wmma_f32_16x16x32_bf16(false, afrag[jm], false, bfrag,
                                                              (short)0, acc[jm][jn], false, false);
    }
  }

  // epilogue: + bias, f32 -> bf16 store. C layout: VGPR r -> M = r + 8*half, N = r16.
  #pragma unroll
  for (int jn = 0; jn < 2; ++jn) {
    const int n = n0 + (wn + jn) * 16 + r16;
    const float bias = bx[n] + bh[n];
    #pragma unroll
    for (int jm = 0; jm < 2; ++jm) {
      #pragma unroll
      for (int r = 0; r < 8; ++r) {
        const int m = m0 + (wm + jm) * 16 + half * 8 + r;
        xp[(size_t)m * HH + n] = f32_to_bf16_rne(acc[jm][jn][r] + bias);
      }
    }
  }
}

// =====================================================================================
// Phase 2: one recurrent step: h_next = tanh(h_prev @ W_h^T + xp_t)
// M = B = 64 (whole batch per block), N tile = 128 (8 blocks), K = H = 1024.
// =====================================================================================
__global__ __launch_bounds__(256)
void rnn_step_kernel(const float* __restrict__ h_prev,  // [B,H] f32
                     const uint16_t* __restrict__ whb,  // [H,H] bf16
                     const uint16_t* __restrict__ xp,   // [B*T,H] bf16
                     float* __restrict__ h_next,        // [B,H] f32
                     int t)
{
  __shared__ uint16_t As[64 * ROWP];
  __shared__ uint16_t Bs[128 * ROWP];

  const int n0   = blockIdx.x * 128;
  const int tid  = threadIdx.x;
  const int lane = tid & 31;
  const int wv   = tid >> 5;
  const int wm   = (wv & 1) * 2;
  const int wn   = (wv >> 1) * 2;

  const int srow = tid >> 2;
  const int scol = (tid & 3) * 8;

  const int half = lane >> 4;
  const int r16  = lane & 15;

  const uint32_t bs_lds_off = (uint32_t)(uintptr_t)(void*)Bs;

  v8f acc[2][2] = {};

  for (int k0 = 0; k0 < HH; k0 += KT) {
    __syncthreads();
    if (wv == 0) {  // TDM pulls the 128x32 W_h tile into padded LDS
      tdm_load_w_tile(whb + (size_t)n0 * HH + k0, bs_lds_off, 128);
    }
    { // stage A: h_prev (f32 -> bf16), 64 x 32 tile
      const float* gp = h_prev + (size_t)srow * HH + (k0 + scol);
      uint16_t*    lp = As + srow * ROWP + scol;
      __builtin_prefetch((const void*)(gp + KT), 0, 3);
      #pragma unroll
      for (int j = 0; j < 8; ++j) lp[j] = f32_to_bf16_rne(gp[j]);
    }
    if (wv == 0) __builtin_amdgcn_s_wait_tensorcnt(0);
    __syncthreads();

    v16bf afrag[2];
    #pragma unroll
    for (int jm = 0; jm < 2; ++jm) {
      const uint16_t* ap = As + ((wm + jm) * 16 + r16) * ROWP + half * 8;
      v8bf a_lo = *(const v8bf*)(const void*)(ap);
      v8bf a_hi = *(const v8bf*)(const void*)(ap + 16);
      afrag[jm] = SHUF16(a_lo, a_hi);
    }
    #pragma unroll
    for (int jn = 0; jn < 2; ++jn) {
      const uint16_t* bp = Bs + ((wn + jn) * 16 + r16) * ROWP + half * 8;
      v8bf b_lo = *(const v8bf*)(const void*)(bp);
      v8bf b_hi = *(const v8bf*)(const void*)(bp + 16);
      v16bf bfrag = SHUF16(b_lo, b_hi);
      #pragma unroll
      for (int jm = 0; jm < 2; ++jm)
        acc[jm][jn] = __builtin_amdgcn_wmma_f32_16x16x32_bf16(false, afrag[jm], false, bfrag,
                                                              (short)0, acc[jm][jn], false, false);
    }
  }

  // epilogue: h = tanh(acc + xp_t)
  #pragma unroll
  for (int jn = 0; jn < 2; ++jn) {
    const int n = n0 + (wn + jn) * 16 + r16;
    #pragma unroll
    for (int jm = 0; jm < 2; ++jm) {
      #pragma unroll
      for (int r = 0; r < 8; ++r) {
        const int b = (wm + jm) * 16 + half * 8 + r;  // batch index (M dim)
        const float xpv = bf16_to_f32(xp[((size_t)b * TT + t) * HH + n]);
        h_next[(size_t)b * HH + n] = tanhf(acc[jm][jn][r] + xpv);
      }
    }
  }
}

// =====================================================================================
// Host side
// =====================================================================================
extern "C" void kernel_launch(void* const* d_in, const int* in_sizes, int n_in,
                              void* d_out, int out_size, void* d_ws, size_t ws_size,
                              hipStream_t stream) {
  (void)in_sizes; (void)n_in; (void)out_size; (void)ws_size;
  const float* x   = (const float*)d_in[0];  // [B,T,I]
  const float* W_h = (const float*)d_in[1];  // [H,H]
  const float* b_h = (const float*)d_in[2];  // [H]
  const float* W_x = (const float*)d_in[3];  // [H,I]
  const float* b_x = (const float*)d_in[4];  // [H]
  float* out = (float*)d_out;                // [B,H]

  // workspace layout
  uint8_t* ws = (uint8_t*)d_ws;
  uint16_t* wxb = (uint16_t*)ws;                                   // 2 MB
  uint16_t* whb = (uint16_t*)(ws + (size_t)2 * 1024 * 1024);       // 2 MB
  uint16_t* xp  = (uint16_t*)(ws + (size_t)4 * 1024 * 1024);       // 64 MB
  float*    h0  = (float*)(ws + (size_t)68 * 1024 * 1024);         // 256 KB
  float*    h1  = h0 + BB * HH;                                    // 256 KB

  // convert weights to bf16, zero h0
  cvt_f32_bf16_kernel<<<(HH * II) / 256, 256, 0, stream>>>(W_x, wxb, HH * II);
  cvt_f32_bf16_kernel<<<(HH * HH) / 256, 256, 0, stream>>>(W_h, whb, HH * HH);
  zero_f32_kernel<<<(BB * HH) / 256, 256, 0, stream>>>(h0, BB * HH);

  // phase 1: big input-projection GEMM (M=32768, N=1024, K=1024)
  dim3 g1(HH / 128, (BB * TT) / 64);
  xproj_kernel<<<g1, 256, 0, stream>>>(x, wxb, b_x, b_h, xp);

  // phase 2: 512 dependent recurrent steps (graph-captured launches)
  float* cur = h0;
  float* nxt = h1;
  for (int t = 0; t < TT; ++t) {
    float* dst = (t == TT - 1) ? out : nxt;
    rnn_step_kernel<<<dim3(HH / 128), 256, 0, stream>>>(cur, whb, xp, dst, t);
    float* tmp = cur; cur = nxt; nxt = tmp;
  }
}
